// WeightNet_39127152066738
// MI455X (gfx1250) — matrix operand order repaired and checked by reading, not compiled, not run on hardware
//
#include <hip/hip_runtime.h>
#include <hip/hip_bf16.h>

typedef __attribute__((ext_vector_type(16))) __bf16 v16bf;
typedef __attribute__((ext_vector_type(8)))  float  v8f;

#define HWD 56
#define PIX 3136          // 56*56
#define CIN 256
#define OCH 256
#define KTOT 2304         // CIN*9
#define BATCH 32
#define BM 128
#define BN 128
#define BK 32
#define SA 40             // LDS row stride (elements): 80B rows -> 16B aligned, conflict-free

union U16 { uint4 q[2]; v16bf v; };

static __device__ __forceinline__ unsigned short f2bf(float f) {
    unsigned int u = __float_as_uint(f);
    unsigned int r = u + 0x7FFFu + ((u >> 16) & 1u);   // round-to-nearest-even
    return (unsigned short)(r >> 16);
}

// ---------------- Kernel 1: global average pool -> gap (B, C) ----------------
__global__ __launch_bounds__(256) void wn_gap(const float* __restrict__ x,
                                              float* __restrict__ gap) {
    __shared__ float red[256];
    const int bc = blockIdx.x;                      // b*256 + c
    const float* xp = x + (size_t)bc * PIX;
    float s = 0.f;
    for (int i = threadIdx.x; i < PIX; i += 256) s += xp[i];
    red[threadIdx.x] = s;
    __syncthreads();
    for (int o = 128; o > 0; o >>= 1) {
        if (threadIdx.x < (unsigned)o) red[threadIdx.x] += red[threadIdx.x + o];
        __syncthreads();
    }
    if (threadIdx.x == 0) gap[bc] = red[0] * (1.f / (float)PIX);
}

// ---------------- Kernel 2: reduce FC + fc1 + sigmoid -> a (B, 512) ----------
__global__ __launch_bounds__(256) void wn_act(const float* __restrict__ gap,
                                              const float* __restrict__ rw,
                                              const float* __restrict__ rb,
                                              const float* __restrict__ f1w,
                                              float* __restrict__ aOut) {
    __shared__ float gL[256];
    __shared__ float gg[16];
    const int b = blockIdx.x;
    gL[threadIdx.x] = gap[b * 256 + threadIdx.x];
    __syncthreads();
    if (threadIdx.x < 16) {
        float s = rb[threadIdx.x];
        for (int c = 0; c < 256; ++c) s += gL[c] * rw[threadIdx.x * 256 + c];
        gg[threadIdx.x] = s;
    }
    __syncthreads();
    for (int m = threadIdx.x; m < 512; m += 256) {
        float s = 0.f;
        #pragma unroll
        for (int j = 0; j < 16; ++j) s += gg[j] * f1w[m * 16 + j];
        aOut[b * 512 + m] = 1.f / (1.f + expf(-s));
    }
}

// ---------------- Kernel 3: per-sample implicit-GEMM conv via WMMA -----------
// Per batch b:  Y[oc,p] = sum_k Wgen[oc,k] * im2col(x)[k,p]
//   Wgen[oc,k] = a[b, 2*oc + (k>=1152)] * fc2w[oc*2304+k] + fc2b[oc*2304+k]
__global__ __launch_bounds__(256) void wn_gemm(const float* __restrict__ x,
                                               const float* __restrict__ fc2w,
                                               const float* __restrict__ fc2b,
                                               const float* __restrict__ aAct,
                                               float* __restrict__ y) {
    __shared__ alignas(16) unsigned short As[BM * SA];   // [m][k] bf16
    __shared__ alignas(16) unsigned short Bs[BN * SA];   // [n][k] bf16 (col-major tile)
    __shared__ float aL[512];

    const int b    = blockIdx.z;
    const int mb   = blockIdx.y * BM;
    const int nb   = blockIdx.x * BN;
    const int t    = threadIdx.x;
    const int lane = t & 31;
    const int wave = t >> 5;
    const int wm   = wave & 3;        // 4 waves along M: 32-row strip each
    const int wn   = wave >> 2;       // 2 waves along N: 64-col strip each

    aL[t]       = aAct[b * 512 + t];
    aL[t + 256] = aAct[b * 512 + 256 + t];
    __syncthreads();

    v8f zero = {0.f, 0.f, 0.f, 0.f, 0.f, 0.f, 0.f, 0.f};
    v8f acc[2][4];
    #pragma unroll
    for (int i = 0; i < 2; ++i)
        #pragma unroll
        for (int j = 0; j < 4; ++j) acc[i][j] = zero;

    const float* xb = x + (size_t)b * CIN * PIX;
    const int half = lane >> 4;
    const int lrow = lane & 15;

    // ---- loop-invariant staging coordinates ----
    // A tile: thread owns K-quad kq (0..7), rows m = i*32 + (t>>3)
    const int kq  = t & 7;
    const int ma0 = t >> 3;            // 0..31
    // B tile: thread owns column n = t&127, K-half th (0/1)
    const int nB  = t & 127;
    const int thB = t >> 7;            // 0 or 1
    const int pB  = nb + nB;
    const int ohB = pB / HWD;
    const int owB = pB - ohB * HWD;
    const bool pok = pB < PIX;

    for (int k0 = 0; k0 < KTOT; k0 += BK) {
        // ---- stage A tile (generated weights), 128 x 32 via float4 loads ----
        {
            const int kg = k0 + kq * 4;          // never straddles 1152 (all mult of 4)
            const int sel = (kg >= 1152) ? 1 : 0;
            #pragma unroll
            for (int i = 0; i < 4; ++i) {
                int m  = i * 32 + ma0;
                int oc = mb + m;
                int f  = oc * KTOT + kg;         // 16B aligned
                float4 w4 = *(const float4*)(fc2w + f);
                float4 b4 = *(const float4*)(fc2b + f);
                float av  = aL[oc * 2 + sel];
                unsigned int lo = (unsigned)f2bf(av * w4.x + b4.x)
                                | ((unsigned)f2bf(av * w4.y + b4.y) << 16);
                unsigned int hi = (unsigned)f2bf(av * w4.z + b4.z)
                                | ((unsigned)f2bf(av * w4.w + b4.w) << 16);
                uint2 pk; pk.x = lo; pk.y = hi;
                *(uint2*)&As[m * SA + kq * 4] = pk;   // 8B aligned
            }
        }
        // ---- stage B tile (im2col of x), 32 x 128 stored [n][k] ----
        {
            #pragma unroll
            for (int j = 0; j < 8; ++j) {
                unsigned int pack = 0;
                #pragma unroll
                for (int s = 0; s < 2; ++s) {
                    int kk = thB * 16 + j * 2 + s;
                    int k  = k0 + kk;
                    int c  = k / 9;
                    int r9 = k - c * 9;
                    int kh = r9 / 3;
                    int kw = r9 - kh * 3;
                    int ih = ohB + kh - 1;
                    int iw = owB + kw - 1;
                    float v = 0.f;
                    if (pok && (unsigned)ih < HWD && (unsigned)iw < HWD)
                        v = xb[c * PIX + ih * HWD + iw];
                    pack |= (unsigned)f2bf(v) << (16 * s);
                }
                *(unsigned int*)&Bs[nB * SA + thB * 16 + j * 2] = pack;  // 4B aligned
            }
        }
        __syncthreads();

        // ---- fragment loads (ISA 7.12.2 bf16 layouts) ----
        U16 aF[2];
        #pragma unroll
        for (int tI = 0; tI < 2; ++tI) {
            const unsigned short* pA = &As[(wm * 32 + tI * 16 + lrow) * SA + half * 8];
            aF[tI].q[0] = *(const uint4*)(pA);        // K = half*8 + 0..7
            aF[tI].q[1] = *(const uint4*)(pA + 16);   // K = 16 + half*8 + 0..7
        }
        U16 bF[4];
        #pragma unroll
        for (int u = 0; u < 4; ++u) {
            const unsigned short* pB = &Bs[(wn * 64 + u * 16 + lrow) * SA + half * 8];
            bF[u].q[0] = *(const uint4*)(pB);         // K = half*8 + 0..7
            bF[u].q[1] = *(const uint4*)(pB + 16);    // K = 16 + half*8 + 0..7
        }

        #pragma unroll
        for (int tI = 0; tI < 2; ++tI)
            #pragma unroll
            for (int u = 0; u < 4; ++u)
                acc[tI][u] = __builtin_amdgcn_wmma_f32_16x16x32_bf16(
                    false, aF[tI].v, false, bF[u].v, (short)0, acc[tI][u],
                    false, false);
        __syncthreads();
    }

    // ---- epilogue: D layout -> row = r + (lane>>4)*8, col = lane&15 ----
    float* yb = y + ((size_t)b * OCH + mb) * PIX;
    #pragma unroll
    for (int tI = 0; tI < 2; ++tI) {
        int ocl = wm * 32 + tI * 16 + half * 8;
        #pragma unroll
        for (int u = 0; u < 4; ++u) {
            int p = nb + wn * 64 + u * 16 + lrow;
            if (p < PIX) {
                #pragma unroll
                for (int r = 0; r < 8; ++r)
                    yb[(size_t)(ocl + r) * PIX + p] = acc[tI][u][r];
            }
        }
    }
}

extern "C" void kernel_launch(void* const* d_in, const int* in_sizes, int n_in,
                              void* d_out, int out_size, void* d_ws, size_t ws_size,
                              hipStream_t stream) {
    const float* x   = (const float*)d_in[0];
    const float* rw  = (const float*)d_in[1];
    const float* rb  = (const float*)d_in[2];
    const float* f1w = (const float*)d_in[3];
    const float* f2w = (const float*)d_in[4];
    const float* f2b = (const float*)d_in[5];
    float* y = (float*)d_out;

    float* gap  = (float*)d_ws;          // 32*256 floats
    float* aAct = gap + BATCH * CIN;     // 32*512 floats (96 KB total ws use)

    wn_gap<<<dim3(BATCH * CIN), 256, 0, stream>>>(x, gap);
    wn_act<<<dim3(BATCH), 256, 0, stream>>>(gap, rw, rb, f1w, aAct);

    dim3 grid((PIX + BN - 1) / BN, OCH / BM, BATCH);   // (25, 2, 32)
    wn_gemm<<<grid, 256, 0, stream>>>(x, f2w, f2b, aAct, y);
}
